// DecLayer_3925600108555
// MI455X (gfx1250) — matrix operand order repaired and checked by reading, not compiled, not run on hardware
//
#include <hip/hip_runtime.h>
#include <hip/hip_bf16.h>
#include <math.h>

typedef __attribute__((ext_vector_type(16))) __bf16 v16bf;
typedef __attribute__((ext_vector_type(8)))  __bf16 v8bf;
typedef __attribute__((ext_vector_type(8)))  float  v8f;

#define NTHREADS 256
#define H_    128
#define HIN_  384
#define CIN_  512
#define KNB_  48
#define DFF_  512
#define NPW1  16      // nodes per workgroup, kernel 1
#define NPW2  32      // nodes per workgroup, kernel 2

__device__ __forceinline__ float gelu_f(float x) {
  return 0.5f * x * (1.0f + erff(x * 0.70710678118654752440f));
}

__device__ __forceinline__ v8f wmma_bf16(v16bf a, v16bf b, v8f c) {
  return __builtin_amdgcn_wmma_f32_16x16x32_bf16(false, a, false, b, (short)0, c,
                                                 false, false);
}

__device__ __forceinline__ v16bf ldfrag(const __bf16* base, int tile, int l) {
  return *(const v16bf*)(base + (tile * 32 + l) * 16);
}

__device__ __forceinline__ v8bf cvt8(float4 a, float4 b) {
  v8bf o;
  o[0] = (__bf16)a.x; o[1] = (__bf16)a.y; o[2] = (__bf16)a.z; o[3] = (__bf16)a.w;
  o[4] = (__bf16)b.x; o[5] = (__bf16)b.y; o[6] = (__bf16)b.z; o[7] = (__bf16)b.w;
  return o;
}

// Map matrix element (row r, k) to staged A-fragment layout
// [r/16][k/32][lane(32)][16 halves] (CDNA5 16-bit A VGPR layout).
__device__ __forceinline__ int a_idx(int r, int k, int kt) {
  int k32 = k & 31;
  int hh  = (k32 >> 3) & 1;
  int j   = (k32 & 7) + ((k32 & 16) >> 1);
  return ((((r >> 4) * kt) + (k >> 5)) * 32 + ((hh << 4) | (r & 15))) * 16 + j;
}

// Stage W[rows=NT*16][K=SK] (f32 row-major; B = W^T) into LDS bf16 B-fragment
// layout [t(KT)][nt(NT)][lane(32)][16]. Each lane's 16 halves are two 8-k runs
// (k0..k0+7, k0+16..k0+23): 4x b128 global loads -> 2x b128 LDS stores.
template<int KT, int NT, int SK>
__device__ __forceinline__ void stage_B(__bf16* dst, const float* __restrict__ src,
                                        int tid) {
  const int groups = KT * NT * 32;
  for (int g = tid; g < groups; g += NTHREADS) {
    int l  = g & 31;
    int nt = (g >> 5) & (NT - 1);
    int t  = g / (32 * NT);
    int k0 = 32 * t + 8 * (l >> 4);
    int c  = 16 * nt + (l & 15);
    const float4* s = (const float4*)(src + c * SK + k0);
    float4 x0 = s[0], x1 = s[1], x2 = s[4], x3 = s[5];
    v8bf* dp = (v8bf*)(dst + g * 16);
    dp[0] = cvt8(x0, x1);
    dp[1] = cvt8(x2, x3);
  }
}

// ---------------------------------------------------------------------------
// Kernel 1: per-edge 3-layer MLP (bf16 WMMA) + masked K-reduction.
// ---------------------------------------------------------------------------
__global__ __launch_bounds__(NTHREADS) void k_edge_message(
    const float* __restrict__ hV, const float* __restrict__ hE,
    const float* __restrict__ maskA,
    const float* __restrict__ W1w, const float* __restrict__ W1b,
    const float* __restrict__ W2w, const float* __restrict__ W2b,
    const float* __restrict__ W3w, const float* __restrict__ W3b,
    float* __restrict__ ws_x)
{
  extern __shared__ __align__(32) char smem[];
  __bf16* sW1 = (__bf16*)smem;               // [16][8][32][16] = 65536 halves
  __bf16* sW2 = sW1 + 16 * 8 * 512;          // [4][8][32][16]  = 16384
  __bf16* sW3 = sW2 + 4 * 8 * 512;           // [4][8][32][16]  = 16384
  __bf16* sX  = sW3 + 4 * 8 * 512;           // [3][16][32][16] = 24576 (48x512 A)
  __bf16* sY  = sX + 3 * 16 * 512;           // [3][4][32][16]  = 6144  (48x128 A)
  float*  sMask = (float*)(sY + 3 * 4 * 512);// 64 floats (48 used)
  float*  sRed  = sMask + 64;                // [8 waves][32 lanes]

  const int tid = threadIdx.x;
  const int l   = tid & 31;
  const int w   = tid >> 5;
  const int h   = l >> 4;
  const int col = 16 * w + (l & 15);

  stage_B<16, 8, 512>(sW1, W1w, tid);
  stage_B<4, 8, 128>(sW2, W2w, tid);
  stage_B<4, 8, 128>(sW3, W3w, tid);

  const float b1 = W1b[col];
  const float b2 = W2b[col];
  const float b3 = W3b[col];

  const int node0 = blockIdx.x * NPW1;
  for (int ni = 0; ni < NPW1; ++ni) {
    const int node = node0 + ni;
    __syncthreads();

    // Stage X = [h_V | h_E] (48x512) as bf16 A fragments, in 8-half groups:
    // each group is 8 k-contiguous source floats (2x b128 loads, 1x b128 store).
    for (int g = tid; g < 3 * 16 * 512 / 8; g += NTHREADS) {   // 3072 groups
      int ll = (g >> 1) & 31;
      int t  = (g >> 6) & 15;
      int mt = g >> 10;
      int r  = 16 * mt + (ll & 15);
      int k0 = 32 * t + 8 * (ll >> 4) + ((g & 1) << 4);
      const float* s = (k0 < H_) ? (hV + node * H_ + k0)
                                 : (hE + (size_t)(node * KNB_ + r) * HIN_ + (k0 - H_));
      float4 x0 = ((const float4*)s)[0];
      float4 x1 = ((const float4*)s)[1];
      *(v8bf*)(sX + g * 8) = cvt8(x0, x1);
    }
    if (tid < KNB_) sMask[tid] = maskA[node * KNB_ + tid];
    __syncthreads();

    // --- GEMM1: Y1 = gelu(X @ W1^T + b1)  (48x512)(512x128) ---
    v8f acc1[3] = {};
    {
      v16bf bf = ldfrag(sW1, w, l);
      v16bf a0 = ldfrag(sX, 0, l);
      v16bf a1 = ldfrag(sX, 16, l);
      v16bf a2 = ldfrag(sX, 32, l);
#pragma unroll
      for (int t = 0; t < 16; ++t) {
        int tn = (t + 1) & 15;                 // t=15 reloads t=0 (discarded)
        v16bf bfn = ldfrag(sW1, tn * 8 + w, l);
        v16bf a0n = ldfrag(sX, 0 * 16 + tn, l);
        v16bf a1n = ldfrag(sX, 1 * 16 + tn, l);
        v16bf a2n = ldfrag(sX, 2 * 16 + tn, l);
        acc1[0] = wmma_bf16(a0, bf, acc1[0]);
        acc1[1] = wmma_bf16(a1, bf, acc1[1]);
        acc1[2] = wmma_bf16(a2, bf, acc1[2]);
        bf = bfn; a0 = a0n; a1 = a1n; a2 = a2n;
      }
    }

    // Overlap: prefetch next node's h_E (576 cachelines) while we compute.
    if (ni + 1 < NPW1) {
      const char* pn = (const char*)(hE + (size_t)(node + 1) * KNB_ * HIN_);
      for (int i = tid; i < (KNB_ * HIN_ * 4) / 128; i += NTHREADS)
        __builtin_prefetch(pn + (size_t)i * 128, 0, 1);
    }

#pragma unroll
    for (int mt = 0; mt < 3; ++mt)
#pragma unroll
      for (int v = 0; v < 8; ++v) {
        int r = 16 * mt + 8 * h + v;
        sY[a_idx(r, col, 4)] = (__bf16)gelu_f(acc1[mt][v] + b1);
      }
    __syncthreads();

    // --- GEMM2: Y2 = gelu(Y1 @ W2^T + b2)  (48x128)(128x128) ---
    v8f acc2[3] = {};
#pragma unroll
    for (int t = 0; t < 4; ++t) {
      v16bf bf = ldfrag(sW2, t * 8 + w, l);
      v16bf a0 = ldfrag(sY, 0 * 4 + t, l);
      v16bf a1 = ldfrag(sY, 1 * 4 + t, l);
      v16bf a2 = ldfrag(sY, 2 * 4 + t, l);
      acc2[0] = wmma_bf16(a0, bf, acc2[0]);
      acc2[1] = wmma_bf16(a1, bf, acc2[1]);
      acc2[2] = wmma_bf16(a2, bf, acc2[2]);
    }
    __syncthreads();  // all Y1 reads complete before overwriting sY
#pragma unroll
    for (int mt = 0; mt < 3; ++mt)
#pragma unroll
      for (int v = 0; v < 8; ++v) {
        int r = 16 * mt + 8 * h + v;
        sY[a_idx(r, col, 4)] = (__bf16)gelu_f(acc2[mt][v] + b2);
      }
    __syncthreads();

    // --- GEMM3: Y3 = Y2 @ W3^T + b3, then masked reduce over 48 edges ---
    v8f acc3[3] = {};
#pragma unroll
    for (int t = 0; t < 4; ++t) {
      v16bf bf = ldfrag(sW3, t * 8 + w, l);
      v16bf a0 = ldfrag(sY, 0 * 4 + t, l);
      v16bf a1 = ldfrag(sY, 1 * 4 + t, l);
      v16bf a2 = ldfrag(sY, 2 * 4 + t, l);
      acc3[0] = wmma_bf16(a0, bf, acc3[0]);
      acc3[1] = wmma_bf16(a1, bf, acc3[1]);
      acc3[2] = wmma_bf16(a2, bf, acc3[2]);
    }
    float p = 0.0f;
#pragma unroll
    for (int mt = 0; mt < 3; ++mt)
#pragma unroll
      for (int v = 0; v < 8; ++v) {
        int r = 16 * mt + 8 * h + v;
        p += sMask[r] * (acc3[mt][v] + b3);
      }
    sRed[w * 32 + l] = p;
    __syncthreads();
    if (tid < H_) {
      int ww = tid >> 4, i = tid & 15;
      float s = sRed[ww * 32 + i] + sRed[ww * 32 + i + 16];
      ws_x[node * H_ + tid] = hV[node * H_ + tid] + s * (1.0f / 30.0f);
    }
  }
}

// ---------------------------------------------------------------------------
// Kernel 2: h = LN1(x); h = LN2(h + Wout(gelu(Win h))); out = mask_V * h
// ---------------------------------------------------------------------------
__device__ __forceinline__ void ln_stats32(const float* sH, float* sRa, float* sRb,
                                           float* sMu, float* sRs, int tid) {
  int n = tid >> 3, c0 = (tid & 7) * 16;
  float s = 0.0f, ss = 0.0f;
  for (int c = 0; c < 16; ++c) {
    float v = sH[n * 128 + c0 + c];
    s += v; ss += v * v;
  }
  sRa[tid] = s; sRb[tid] = ss;
  __syncthreads();
  if (tid < NPW2) {
    float s8 = 0.0f, ss8 = 0.0f;
    for (int i = 0; i < 8; ++i) { s8 += sRa[tid * 8 + i]; ss8 += sRb[tid * 8 + i]; }
    float mu  = s8 * (1.0f / 128.0f);
    float var = ss8 * (1.0f / 128.0f) - mu * mu;
    sMu[tid] = mu;
    sRs[tid] = rsqrtf(var + 1e-5f);
  }
  __syncthreads();
}

__global__ __launch_bounds__(NTHREADS) void k_node_update(
    const float* __restrict__ ws_x, const float* __restrict__ maskV,
    const float* __restrict__ ln1w, const float* __restrict__ ln1b,
    const float* __restrict__ Winw, const float* __restrict__ Winb,
    const float* __restrict__ Woutw, const float* __restrict__ Woutb,
    const float* __restrict__ ln2w, const float* __restrict__ ln2b,
    float* __restrict__ out)
{
  extern __shared__ __align__(32) char smem[];
  __bf16* sW  = (__bf16*)smem;          // 65536 halves: Win, then Wout (reused)
  __bf16* sHA = sW + 16 * 8 * 512;      // [2][4][32][16]  = 4096  (32x128 A)
  __bf16* sG  = sHA + 2 * 4 * 512;      // [2][16][32][16] = 16384 (32x512 A)
  float*  sH  = (float*)(sG + 2 * 16 * 512);  // 32x128 f32
  float*  sRa = sH + NPW2 * 128;        // 256
  float*  sRb = sRa + 256;              // 256
  float*  sMu = sRb + 256;              // 32
  float*  sRs = sMu + NPW2;             // 32

  const int tid = threadIdx.x;
  const int l = tid & 31, w = tid >> 5, h = l >> 4;
  const int node0 = blockIdx.x * NPW2;

  stage_B<4, 32, 128>(sW, Winw, tid);

  for (int g = tid; g < NPW2 * 128 / 4; g += NTHREADS)
    ((float4*)sH)[g] = ((const float4*)(ws_x + node0 * 128))[g];
  __syncthreads();

  // LN1
  ln_stats32(sH, sRa, sRb, sMu, sRs, tid);
  for (int d = tid; d < NPW2 * 128; d += NTHREADS) {
    int n = d >> 7, c = d & 127;
    float hv = (sH[d] - sMu[n]) * sRs[n] * ln1w[c] + ln1b[c];
    sH[d] = hv;                          // f32 h kept for residual
    sHA[a_idx(n, c, 4)] = (__bf16)hv;    // bf16 A operand
  }
  __syncthreads();  // also covers Win staging

  // --- GEMM1: G = gelu(h @ Win^T + b_in)  (32x128)(128x512) ---
  for (int g = 0; g < 4; ++g) {
    int nt = w + 8 * g;
    int cg = 16 * nt + (l & 15);
    float bi = Winb[cg];
    v8f acc[2] = {};
#pragma unroll
    for (int t = 0; t < 4; ++t) {
      v16bf bf = ldfrag(sW, t * 32 + nt, l);
      v16bf a0 = ldfrag(sHA, 0 * 4 + t, l);
      v16bf a1 = ldfrag(sHA, 1 * 4 + t, l);
      acc[0] = wmma_bf16(a0, bf, acc[0]);
      acc[1] = wmma_bf16(a1, bf, acc[1]);
    }
#pragma unroll
    for (int mt = 0; mt < 2; ++mt)
#pragma unroll
      for (int v = 0; v < 8; ++v) {
        int r = 16 * mt + 8 * h + v;
        sG[a_idx(r, cg, 16)] = (__bf16)gelu_f(acc[mt][v] + bi);
      }
  }
  __syncthreads();
  stage_B<16, 8, 512>(sW, Woutw, tid);
  __syncthreads();

  // --- GEMM2: dh = G @ Wout^T + b_out (32x512)(512x128); residual in place ---
  {
    int cg = 16 * w + (l & 15);
    float bo = Woutb[cg];
    v8f acc[2] = {};
    v16bf bf = ldfrag(sW, w, l);
    v16bf a0 = ldfrag(sG, 0, l);
    v16bf a1 = ldfrag(sG, 16, l);
#pragma unroll
    for (int t = 0; t < 16; ++t) {
      int tn = (t + 1) & 15;
      v16bf bfn = ldfrag(sW, tn * 8 + w, l);
      v16bf a0n = ldfrag(sG, 0 * 16 + tn, l);
      v16bf a1n = ldfrag(sG, 1 * 16 + tn, l);
      acc[0] = wmma_bf16(a0, bf, acc[0]);
      acc[1] = wmma_bf16(a1, bf, acc[1]);
      bf = bfn; a0 = a0n; a1 = a1n;
    }
#pragma unroll
    for (int mt = 0; mt < 2; ++mt)
#pragma unroll
      for (int v = 0; v < 8; ++v) {
        int r = 16 * mt + 8 * h + v;
        sH[r * 128 + cg] += acc[mt][v] + bo;   // unique (r,cg) owner
      }
  }
  __syncthreads();

  // LN2 + mask + store
  ln_stats32(sH, sRa, sRb, sMu, sRs, tid);
  for (int d = tid; d < NPW2 * 128; d += NTHREADS) {
    int n = d >> 7, c = d & 127;
    int node = node0 + n;
    float v = (sH[d] - sMu[n]) * sRs[n] * ln2w[c] + ln2b[c];
    out[node * 128 + c] = maskV[node] * v;
  }
}

// ---------------------------------------------------------------------------
extern "C" void kernel_launch(void* const* d_in, const int* in_sizes, int n_in,
                              void* d_out, int out_size, void* d_ws, size_t ws_size,
                              hipStream_t stream) {
  const float* hV    = (const float*)d_in[0];
  const float* hE    = (const float*)d_in[1];
  const float* maskV = (const float*)d_in[2];
  const float* maskA = (const float*)d_in[3];
  const float* W1w   = (const float*)d_in[4];
  const float* W1b   = (const float*)d_in[5];
  const float* W2w   = (const float*)d_in[6];
  const float* W2b   = (const float*)d_in[7];
  const float* W3w   = (const float*)d_in[8];
  const float* W3b   = (const float*)d_in[9];
  const float* ln1w  = (const float*)d_in[10];
  const float* ln1b  = (const float*)d_in[11];
  const float* Winw  = (const float*)d_in[12];
  const float* Winb  = (const float*)d_in[13];
  const float* Woutw = (const float*)d_in[14];
  const float* Woutb = (const float*)d_in[15];
  const float* ln2w  = (const float*)d_in[16];
  const float* ln2b  = (const float*)d_in[17];
  float* out = (float*)d_out;
  float* ws  = (float*)d_ws;   // [nodes][128] f32 = 4 MB

  const int nodes = in_sizes[0] / H_;   // B*N = 8192

  const size_t lds1 =
      (size_t)(16 * 8 * 512 + 4 * 8 * 512 + 4 * 8 * 512 + 3 * 16 * 512 +
               3 * 4 * 512) * 2 +
      (size_t)(64 + 8 * 32) * sizeof(float);          // ~259 KB (<320KB/WGP)
  k_edge_message<<<nodes / NPW1, NTHREADS, lds1, stream>>>(
      hV, hE, maskA, W1w, W1b, W2w, W2b, W3w, W3b, ws);

  const size_t lds2 =
      (size_t)(16 * 8 * 512 + 2 * 4 * 512 + 2 * 16 * 512) * 2 +
      (size_t)(NPW2 * 128 + 256 + 256 + NPW2 + NPW2) * sizeof(float); // ~191 KB
  k_node_update<<<nodes / NPW2, NTHREADS, lds2, stream>>>(
      ws, maskV, ln1w, ln1b, Winw, Winb, Woutw, Woutb, ln2w, ln2b, out);
}
